// Head_18150531793541
// MI455X (gfx1250) — matrix operand order repaired and checked by reading, not compiled, not run on hardware
//
#include <hip/hip_runtime.h>

typedef __attribute__((ext_vector_type(16))) _Float16 v16h;
typedef __attribute__((ext_vector_type(8)))  float    v8f;

constexpr int BB = 4, TT = 4096, CC = 1024, HD = 64;
constexpr float SCL = 1.0f / 32.0f;   // 1/sqrt(C)

// A-matrix (16-bit, 16x32) per-lane K index for element e, lane-half kh:
//   K = e + 16*kh (contiguous, matches B operand fills)
__device__ __forceinline__ int a_k(int e, int kh) {
    return e + kh * 16;
}

// ---------------------------------------------------------------------------
// Swizzle Wq/Wk/Wv [C,64] f32 -> WMMA B-layout f16 tiles.
// wb[((w*4+n16)*32+kb)*512 + lane*16 + e] = W[kb*32 + (lane/16)*16 + e][n16*16 + lane%16]
// ---------------------------------------------------------------------------
__global__ __launch_bounds__(32) void swz_w(const float* __restrict__ Wq,
                                            const float* __restrict__ Wk,
                                            const float* __restrict__ Wv,
                                            _Float16* __restrict__ wb) {
    int blk  = blockIdx.x;           // 0..383 = w*128 + n16*32 + kb
    int lane = threadIdx.x;
    int kb   = blk & 31;
    int n16  = (blk >> 5) & 3;
    int w    = blk >> 7;
    const float* W = (w == 0) ? Wq : (w == 1) ? Wk : Wv;
    int h     = n16 * 16 + (lane & 15);
    int cbase = kb * 32 + (lane >> 4) * 16;
    _Float16* dst = wb + (size_t)blk * 512 + lane * 16;
#pragma unroll
    for (int e = 0; e < 16; ++e)
        dst[e] = (_Float16)W[(size_t)(cbase + e) * HD + h];
}

// ---------------------------------------------------------------------------
// Projection: per wave, one 16-row tile of x -> q,k (row-major f16) and
// v transposed vt[b][h][t] f16. 12 WMMA accumulators over K=1024.
// ---------------------------------------------------------------------------
__global__ __launch_bounds__(32) void proj(const float* __restrict__ x,
                                           const _Float16* __restrict__ wb,
                                           _Float16* __restrict__ qh,
                                           _Float16* __restrict__ kh,
                                           _Float16* __restrict__ vt) {
    int tile = blockIdx.x;            // 0..1023
    int lane = threadIdx.x;
    int row0 = tile * 16;             // flat row in [0, B*T)
    int lm = lane & 15, lh = lane >> 4;

    v8f acc[3][4];
#pragma unroll
    for (int w = 0; w < 3; ++w)
#pragma unroll
        for (int n = 0; n < 4; ++n) acc[w][n] = (v8f){};

    const float* xrow = x + (size_t)(row0 + lm) * CC;
    for (int kb = 0; kb < 32; ++kb) {
        v16h a;
#pragma unroll
        for (int e = 0; e < 16; ++e)
            a[e] = (_Float16)xrow[kb * 32 + a_k(e, lh)];
#pragma unroll
        for (int w = 0; w < 3; ++w) {
#pragma unroll
            for (int n = 0; n < 4; ++n) {
                const v16h* bp = (const v16h*)(wb + (size_t)((w * 4 + n) * 32 + kb) * 512);
                v16h bt = bp[lane];
                acc[w][n] = __builtin_amdgcn_wmma_f32_16x16x32_f16(
                    false, a, false, bt, (short)0, acc[w][n], false, false);
            }
        }
    }
    // Epilogue: C-layout element r -> row r + 8*lh, col n*16 + lm
#pragma unroll
    for (int n = 0; n < 4; ++n) {
#pragma unroll
        for (int r = 0; r < 8; ++r) {
            int m   = r + 8 * lh;
            int col = n * 16 + lm;
            size_t g = (size_t)(row0 + m);
            qh[g * HD + col] = (_Float16)acc[0][n][r];
            kh[g * HD + col] = (_Float16)acc[1][n][r];
            int bb = (int)(g >> 12);       // / 4096
            int t  = (int)(g & 4095);
            vt[((size_t)bb * HD + col) * TT + t] = (_Float16)acc[2][n][r];
        }
    }
}

// ---------------------------------------------------------------------------
// Flash attention (causal), one wave per 16-row Q tile, 32-key chunks.
// ---------------------------------------------------------------------------
__global__ __launch_bounds__(32) void attn(const _Float16* __restrict__ qh,
                                           const _Float16* __restrict__ kh,
                                           const _Float16* __restrict__ vt,
                                           float* __restrict__ out) {
    __shared__ _Float16 Pst[16 * 34];   // padded stride 34 -> conflict-free transpose
    int blk  = blockIdx.x;              // 0..1023
    int lane = threadIdx.x;
    int b  = blk >> 8;
    int qi = blk & 255;
    int q0 = qi * 16;                   // row base within batch
    int lm = lane & 15, lh = lane >> 4;

    // Q in A-layout (two K-tiles: h 0..31 and 32..63)
    const _Float16* qrow = qh + (size_t)(b * TT + q0 + lm) * HD;
    v16h aq0, aq1;
#pragma unroll
    for (int e = 0; e < 16; ++e) {
        int k = a_k(e, lh);
        aq0[e] = qrow[k];
        aq1[e] = qrow[32 + k];
    }

    float mrun[8], lrun[8];
    v8f o[4];
#pragma unroll
    for (int r = 0; r < 8; ++r) { mrun[r] = -__builtin_inff(); lrun[r] = 0.0f; }
#pragma unroll
    for (int t = 0; t < 4; ++t) o[t] = (v8f){};

    int nch = (q0 + 16 + 31) >> 5;
    for (int ch = 0; ch < nch; ++ch) {
        int kc = ch * 32;
        // ---- S = Q K^T : B-layout K tiles are contiguous v16h loads ----
        const _Float16* kb0 = kh + (size_t)(b * TT + kc + lm) * HD + lh * 16;
        const _Float16* kb1 = kb0 + 16 * HD;
        v16h b0lo = *(const v16h*)(kb0);
        v16h b0hi = *(const v16h*)(kb0 + 32);
        v16h b1lo = *(const v16h*)(kb1);
        v16h b1hi = *(const v16h*)(kb1 + 32);
        v8f zero = {};
        v8f s0 = __builtin_amdgcn_wmma_f32_16x16x32_f16(false, aq0, false, b0lo, (short)0, zero, false, false);
        s0     = __builtin_amdgcn_wmma_f32_16x16x32_f16(false, aq1, false, b0hi, (short)0, s0,   false, false);
        v8f s1 = __builtin_amdgcn_wmma_f32_16x16x32_f16(false, aq0, false, b1lo, (short)0, zero, false, false);
        s1     = __builtin_amdgcn_wmma_f32_16x16x32_f16(false, aq1, false, b1hi, (short)0, s1,   false, false);

        // ---- online softmax: rows live across 16-lane halves ----
        float alpha[8];
#pragma unroll
        for (int r = 0; r < 8; ++r) {
            int qg = q0 + r + 8 * lh;
            int k0 = kc + lm, k1 = k0 + 16;
            float v0 = (k0 <= qg) ? s0[r] * SCL : -__builtin_inff();
            float v1 = (k1 <= qg) ? s1[r] * SCL : -__builtin_inff();
            float mx = fmaxf(v0, v1);
#pragma unroll
            for (int msk = 1; msk <= 8; msk <<= 1) mx = fmaxf(mx, __shfl_xor(mx, msk, 32));
            float mnew = fmaxf(mrun[r], mx);
            float a  = __expf(mrun[r] - mnew);
            float p0 = __expf(v0 - mnew);
            float p1 = __expf(v1 - mnew);
            float rs = p0 + p1;
#pragma unroll
            for (int msk = 1; msk <= 8; msk <<= 1) rs += __shfl_xor(rs, msk, 32);
            lrun[r] = lrun[r] * a + rs;
            mrun[r] = mnew;
            alpha[r] = a;
            int mrow = r + 8 * lh;
            Pst[mrow * 34 + lm]      = (_Float16)p0;   // C-layout -> row-major LDS
            Pst[mrow * 34 + 16 + lm] = (_Float16)p1;
        }
#pragma unroll
        for (int t = 0; t < 4; ++t)
#pragma unroll
            for (int r = 0; r < 8; ++r) o[t][r] *= alpha[r];

        __syncthreads();
        v16h ap;                         // read P back in A-layout
#pragma unroll
        for (int e = 0; e < 16; ++e)
            ap[e] = Pst[lm * 34 + a_k(e, lh)];
        __syncthreads();

        // ---- O += P V : vt is [b][h][t] so B tiles are contiguous ----
#pragma unroll
        for (int t = 0; t < 4; ++t) {
            const v16h* vp = (const v16h*)(vt + (size_t)(b * HD + t * 16 + lm) * TT + kc + lh * 16);
            v16h bv = *vp;
            o[t] = __builtin_amdgcn_wmma_f32_16x16x32_f16(false, ap, false, bv, (short)0, o[t], false, false);
        }
    }

    // normalize and store f32 output [B,T,64]
#pragma unroll
    for (int t = 0; t < 4; ++t)
#pragma unroll
        for (int r = 0; r < 8; ++r) {
            int m = r + 8 * lh;
            out[(size_t)(b * TT + q0 + m) * HD + t * 16 + lm] = o[t][r] / lrun[r];
        }
}

extern "C" void kernel_launch(void* const* d_in, const int* in_sizes, int n_in,
                              void* d_out, int out_size, void* d_ws, size_t ws_size,
                              hipStream_t stream) {
    const float* x  = (const float*)d_in[0];
    const float* Wq = (const float*)d_in[1];
    const float* Wk = (const float*)d_in[2];
    const float* Wv = (const float*)d_in[3];
    float* out = (float*)d_out;

    char* ws = (char*)d_ws;
    _Float16* qh = (_Float16*)(ws);                               // 2 MB
    _Float16* kh = (_Float16*)(ws + (size_t)2 * 1024 * 1024);     // 2 MB
    _Float16* vt = (_Float16*)(ws + (size_t)4 * 1024 * 1024);     // 2 MB
    _Float16* wb = (_Float16*)(ws + (size_t)6 * 1024 * 1024);     // 384 KB

    swz_w<<<384,  32, 0, stream>>>(Wq, Wk, Wv, wb);
    proj <<<1024, 32, 0, stream>>>(x, wb, qh, kh, vt);
    attn <<<1024, 32, 0, stream>>>(qh, kh, vt, out);
}
